// FastALIFCell_50242527428964
// MI455X (gfx1250) — matrix operand order repaired and checked by reading, not compiled, not run on hardware
//
#include <hip/hip_runtime.h>
#include <hip/hip_fp16.h>

#define UNITS  2048
#define KNZ    204
#define TSTEPS 5
#define BATCH  32
#define BT     (BATCH * TSTEPS)          // 160
#define BTU    (BATCH * TSTEPS * UNITS)  // 327680

#define DECAY   0.9512294245007140f      // exp(-1/20)
#define DECAY_A 0.9950124791926823f      // exp(-1/200)

typedef __attribute__((ext_vector_type(16))) _Float16 v16h;
typedef __attribute__((ext_vector_type(8)))  _Float16 v8h;
typedef __attribute__((ext_vector_type(8)))  float    v8f;

// ---------------------------------------------------------------------------
// Kernel 1: zero dense weight matrix (f16, unit-major) and convert spike
// buffer z_buf0 (B*T*U f32, values 0/1 -> exact in f16).
// ---------------------------------------------------------------------------
__global__ __launch_bounds__(256) void alif_prep_kernel(
    const float* __restrict__ z_buf0, _Float16* __restrict__ Wt,
    _Float16* __restrict__ Zh) {
  int idx = blockIdx.x * blockDim.x + threadIdx.x;
  // Zero Wt: UNITS*UNITS halves = 2,097,152 dwords.
  if (idx < (UNITS * UNITS) / 2) ((unsigned int*)Wt)[idx] = 0u;
  if (idx < BTU) Zh[idx] = (_Float16)z_buf0[idx];
}

// ---------------------------------------------------------------------------
// Kernel 2: scatter sparse weights into dense Wt[u][c] (unit-major).
// cols rows are sorted, so duplicates are adjacent: run-length accumulate in
// f32, one f16 store per distinct column. Each thread owns one unit row ->
// no atomics, fully deterministic.
// ---------------------------------------------------------------------------
__global__ __launch_bounds__(256) void alif_scatter_w_kernel(
    const float* __restrict__ w_vals, const int* __restrict__ cols,
    _Float16* __restrict__ Wt) {
  int u = blockIdx.x * blockDim.x + threadIdx.x;
  if (u >= UNITS) return;
  const float* w = w_vals + (size_t)u * KNZ;
  const int*   c = cols   + (size_t)u * KNZ;
  _Float16* row = Wt + (size_t)u * UNITS;
  int k = 0;
  while (k < KNZ) {
    int cc = c[k];
    float acc = 0.0f;
    do { acc += w[k]; ++k; } while (k < KNZ && c[k] == cc);
    row[cc] = (_Float16)acc;
  }
}

// ---------------------------------------------------------------------------
// Kernel 3: i_rec = Zh(160x2048) x Wt^T  via V_WMMA_F32_16X16X32_F16.
// One wave per 16x16 output tile; 64 WMMA ops along K=2048.
// Fragment packing: contiguous k = 16*hi + s in both A and B (wave32).
// ---------------------------------------------------------------------------
__global__ __launch_bounds__(256) void alif_gemm_kernel(
    const _Float16* __restrict__ Zh, const _Float16* __restrict__ Wt,
    float* __restrict__ irec) {
  const int wave = threadIdx.x >> 5;
  const int lane = threadIdx.x & 31;
  const int tile = blockIdx.x * (blockDim.x >> 5) + wave;  // 0..1279 exact
  const int NTN  = UNITS / 16;                             // 128 N-tiles
  const int mtile = tile / NTN;                            // 0..9
  const int ntile = tile % NTN;

  const int m  = lane & 15;
  const int hi = lane >> 4;
  const int n  = lane & 15;

  const _Float16* zrow = Zh + (size_t)(mtile * 16 + m) * UNITS;
  const _Float16* wrow = Wt + (size_t)(ntile * 16 + n) * UNITS;

  v8f acc = {};
  for (int k0 = 0; k0 < UNITS; k0 += 32) {
    // A 16x32: lane(m,hi): halves 0..7 -> K=hi*16+p ; 8..15 -> K=hi*16+8+p
    v8h alo = *(const v8h*)(zrow + k0 + hi * 16);
    v8h ahi = *(const v8h*)(zrow + k0 + hi * 16 + 8);
    v16h a;
#pragma unroll
    for (int p = 0; p < 8; ++p) { a[p] = alo[p]; a[p + 8] = ahi[p]; }
    // B 32x16: lane(n,hi): half p -> (K = hi*16+p, N = n), contiguous in Wt.
    v16h b = *(const v16h*)(wrow + k0 + hi * 16);
    acc = __builtin_amdgcn_wmma_f32_16x16x32_f16(
        false, a, false, b, (short)0, acc, false, false);
  }
  // D 16x16 f32: VGPR r -> M = r + 8*hi, N = n.
  float* out = irec + (size_t)(mtile * 16) * UNITS + (size_t)ntile * 16 + n;
#pragma unroll
  for (int r = 0; r < 8; ++r) out[(size_t)(r + 8 * hi) * UNITS] = acc[r];
}

// ---------------------------------------------------------------------------
// Kernel 4: pointwise ALIF dynamics per (b,u); all T=5 filters as scalar
// recurrences. Outputs concatenated: [z | v_buf | a_buf | v_sc].
// ---------------------------------------------------------------------------
__global__ __launch_bounds__(256) void alif_point_kernel(
    const float* __restrict__ inputs, const float* __restrict__ v0,
    const float* __restrict__ a0, const float* __restrict__ z_buf0,
    const float* __restrict__ beta, const float* __restrict__ irec,
    float* __restrict__ out) {
  int idx = blockIdx.x * blockDim.x + threadIdx.x;  // b*UNITS + u
  if (idx >= BATCH * UNITS) return;
  const int b = idx / UNITS;
  const int u = idx - b * UNITS;

  const float vv0 = v0[idx];
  const float aa0 = a0[idx];
  const float bet = beta[u];

  float it[TSTEPS], oz[TSTEPS];
#pragma unroll
  for (int t = 0; t < TSTEPS; ++t) {
    size_t off = (size_t)b * TSTEPS * UNITS + (size_t)t * UNITS + u;
    it[t] = inputs[off] + irec[off];
    oz[t] = z_buf0[off];
  }

  float tv[TSTEPS], ta[TSTEPS], vsc[TSTEPS], zraw[TSTEPS];
  float fi = 0.0f, pv = DECAY, pa = DECAY_A;
#pragma unroll
  for (int t = 0; t < TSTEPS; ++t) {
    fi = fi * DECAY + it[t];            // filtered_i (Mv causal filter)
    tv[t] = pv * vv0 + fi;              // tilde_v
    ta[t] = pa * aa0;                   // tilde_a
    vsc[t] = tv[t] - bet * ta[t] - 1.0f;  // THR = 1
    zraw[t] = (vsc[t] > 0.0f) ? 1.0f : 0.0f;
    pv *= DECAY;
    pa *= DECAY_A;
  }

  // ref_old[t]: any old spike strictly after t; ref_new[t]: any raw new
  // spike strictly before t.
  bool refOld[TSTEPS];
  float suf = 0.0f;
#pragma unroll
  for (int t = TSTEPS - 1; t >= 0; --t) {
    refOld[t] = (suf > 0.5f);
    suf += oz[t];
  }
  float zm[TSTEPS];
  float pre = 0.0f;
#pragma unroll
  for (int t = 0; t < TSTEPS; ++t) {
    bool refNew = (pre > 0.5f);
    pre += zraw[t];
    zm[t] = (refNew || refOld[t]) ? 0.0f : zraw[t];
  }

  float fr = 0.0f, fa = 0.0f;
#pragma unroll
  for (int t = 0; t < TSTEPS; ++t) {
    fr = fr * DECAY + zm[t];            // filtered reset (THR=1)
    fa = fa * DECAY_A + zm[t];          // Ma filter = (1/DECAY_A)*causal
    size_t off = (size_t)b * TSTEPS * UNITS + (size_t)t * UNITS + u;
    out[off]           = zm[t];
    out[BTU + off]     = tv[t] - fr;               // v_buf
    out[2 * BTU + off] = ta[t] + fa / DECAY_A;     // a_buf
    out[3 * BTU + off] = vsc[t];
  }
}

// ---------------------------------------------------------------------------
extern "C" void kernel_launch(void* const* d_in, const int* in_sizes, int n_in,
                              void* d_out, int out_size, void* d_ws,
                              size_t ws_size, hipStream_t stream) {
  (void)in_sizes; (void)n_in; (void)out_size; (void)ws_size;
  const float* inputs = (const float*)d_in[0];
  const float* v0     = (const float*)d_in[1];
  const float* a0     = (const float*)d_in[2];
  const float* z_buf0 = (const float*)d_in[3];
  const float* w_vals = (const float*)d_in[4];
  const float* beta   = (const float*)d_in[5];
  const int*   cols   = (const int*)d_in[6];
  float* out = (float*)d_out;

  char* ws = (char*)d_ws;
  _Float16* Wt = (_Float16*)ws;                                   // 8 MB
  _Float16* Zh = (_Float16*)(ws + (size_t)UNITS * UNITS * 2);     // 640 KB
  float* irec  = (float*)(ws + (size_t)UNITS * UNITS * 2 +
                               (size_t)BT * UNITS * 2);           // 1.25 MB

  // 1) zero Wt + convert spikes to f16 (covers max(2097152, 327680) ids)
  alif_prep_kernel<<<(UNITS * UNITS / 2 + 255) / 256, 256, 0, stream>>>(
      z_buf0, Wt, Zh);
  // 2) scatter sparse weights into dense f16 Wt[u][c]
  alif_scatter_w_kernel<<<(UNITS + 255) / 256, 256, 0, stream>>>(w_vals, cols,
                                                                 Wt);
  // 3) i_rec GEMM: 1280 tiles, 8 waves/block -> exactly 160 blocks
  alif_gemm_kernel<<<(BT / 16) * (UNITS / 16) / 8, 256, 0, stream>>>(Zh, Wt,
                                                                     irec);
  // 4) pointwise ALIF dynamics + outputs
  alif_point_kernel<<<(BATCH * UNITS + 255) / 256, 256, 0, stream>>>(
      inputs, v0, a0, z_buf0, beta, irec, out);
}
